// MultiHeadAttention_41042707480588
// MI455X (gfx1250) — matrix operand order repaired and checked by reading, not compiled, not run on hardware
//
#include <hip/hip_runtime.h>
#include <hip/hip_bf16.h>
#include <math.h>

// MHA forward for MI455X (gfx1250), wave32, WMMA f16->f32.
// B=2, S=2048, D=1024, H=16, DEPTH=64.
// Fused scores+softmax+PV: the 537MB weights tensor is written exactly once.

typedef __attribute__((ext_vector_type(16))) _Float16 v16h;
typedef __attribute__((ext_vector_type(8)))  _Float16 v8h;
typedef __attribute__((ext_vector_type(8)))  float    v8f;

#define BB 2
#define SS 2048
#define DD 1024
#define HH 16
#define DEPTH 64
#define LDS_P 2056   // padded row stride (floats) to break bank conflicts

// ---------------------------------------------------------------------------
// WMMA wrapper: D = A(16x32 f16) x B(32x16 f16) + C(16x16 f32)
// ---------------------------------------------------------------------------
__device__ __forceinline__ v8f wmma_f16(v16h a, v16h b, v8f c) {
  return __builtin_amdgcn_wmma_f32_16x16x32_f16(
      /*neg_a=*/false, a, /*neg_b=*/false, b,
      /*c_mod=*/(short)0, c, /*reuse_a=*/false, /*reuse_b=*/false);
}

// A-fragment: 16x32 tile of row-major f16 matrix, top-left (row0, k0).
// ISA layout: lanes 0-15 (row m=lane): K = {k0+0..7, k0+16..23}
//             lanes 16-31 (row m=lane-16): K = {k0+8..15, k0+24..31}
__device__ __forceinline__ v16h load_a_f16(const _Float16* __restrict__ A,
                                           int lda, int row0, int k0) {
  int lane = threadIdx.x & 31;
  int hl = lane >> 4;
  int m = lane & 15;
  const _Float16* p = A + (size_t)(row0 + m) * lda + k0 + 8 * hl;
  union { v16h v; v8h h[2]; } u;
  u.h[0] = *(const v8h*)p;          // 16B load
  u.h[1] = *(const v8h*)(p + 16);   // 16B load
  return u.v;
}

// A-fragment from an f32 LDS score strip (row stride LDS_P), converted to f16.
__device__ __forceinline__ v16h load_a_lds_f32(const float* sc, int k0) {
  int lane = threadIdx.x & 31;
  int hl = lane >> 4;
  int m = lane & 15;
  const float* p = sc + (size_t)m * LDS_P + k0 + 8 * hl;
  v16h r;
#pragma unroll
  for (int i = 0; i < 8; ++i) r[i] = (_Float16)p[i];
#pragma unroll
  for (int i = 0; i < 8; ++i) r[8 + i] = (_Float16)p[16 + i];
  return r;
}

// B-fragment: 32x16 tile (KxN) given B stored transposed/row-major as BT[N][K].
// ISA layout: lanes 0-15 (col n=lane): K = {k0+0..7, k0+16..23} ; lanes 16-31: K = {k0+8..15, k0+24..31}
__device__ __forceinline__ v16h load_bt_f16(const _Float16* __restrict__ BT,
                                            int ldb, int n0, int k0) {
  int lane = threadIdx.x & 31;
  int hl = lane >> 4;
  int n = lane & 15;
  const _Float16* p = BT + (size_t)(n0 + n) * ldb + k0 + 8 * hl;
  union { v16h v; v8h h[2]; } u;
  u.h[0] = *(const v8h*)p;          // 16B load
  u.h[1] = *(const v8h*)(p + 16);   // 16B load
  return u.v;
}

// ---------------------------------------------------------------------------
// Kernel 0a: elementwise f32 -> f16 convert
// ---------------------------------------------------------------------------
__global__ __launch_bounds__(256) void cvt_f16_kernel(
    const float* __restrict__ src, _Float16* __restrict__ dst, int n) {
  int i = blockIdx.x * 256 + threadIdx.x;
  if (i < n) dst[i] = (_Float16)src[i];
}

// Kernel 0b: 1024x1024 f32 weight -> f16 transposed (WT[n][k] = W[k][n])
__global__ __launch_bounds__(256) void tcvt_f16_kernel(
    const float* __restrict__ w, _Float16* __restrict__ wT) {
  int i = blockIdx.x * 256 + threadIdx.x;   // 0 .. 1M-1
  int k = i >> 10;
  int n = i & 1023;
  wT[(size_t)n * 1024 + k] = (_Float16)w[i];
}

// ---------------------------------------------------------------------------
// Kernel 1: projection GEMM  Y = Xh @ WhT^T + b  (M=4096, N=1024, K=1024)
// Block: 128 threads = 4 waves stacked on M; each wave computes a 16x64 strip.
// vmode 0: write head-split  [B,H,S,64]   (for q, k)
// vmode 1: write transposed  [B,H,64,S]   (for v -> contiguous B-frags in PV)
// ---------------------------------------------------------------------------
__global__ __launch_bounds__(128) void proj_kernel(
    const _Float16* __restrict__ X, const _Float16* __restrict__ WT,
    const float* __restrict__ bias, _Float16* __restrict__ dst, int vmode) {
  int w = threadIdx.x >> 5;
  int lane = threadIdx.x & 31;
  int idx = lane & 15;
  int hl = lane >> 4;
  int row0 = blockIdx.x * 64 + w * 16;
  int col0 = blockIdx.y * 64;

  v8f acc[4] = {};
  for (int k0 = 0; k0 < DD; k0 += 32) {
    v16h a = load_a_f16(X, DD, row0, k0);
#pragma unroll
    for (int t = 0; t < 4; ++t) {
      v16h bf = load_bt_f16(WT, DD, col0 + 16 * t, k0);
      acc[t] = wmma_f16(a, bf, acc[t]);
    }
  }

#pragma unroll
  for (int t = 0; t < 4; ++t) {
    int c = col0 + 16 * t + idx;
    float bv = bias[c];
    int h = c >> 6;          // head
    int dcol = c & 63;       // col within head
#pragma unroll
    for (int i = 0; i < 8; ++i) {
      int r = row0 + 8 * hl + i;
      int bat = r >> 11;     // r / S
      int s = r & (SS - 1);
      float val = acc[t][i] + bv;
      size_t o;
      if (vmode == 0)
        o = (((size_t)(bat * HH + h) * SS) + s) * DEPTH + dcol;
      else
        o = (((size_t)(bat * HH + h) * DEPTH) + dcol) * SS + s;
      dst[o] = (_Float16)val;
    }
  }
}

// ---------------------------------------------------------------------------
// Kernel 2 (fused): per (head, 16-query strip):
//   scores(16x2048) via WMMA -> LDS -> row softmax in LDS (weights written
//   once, coalesced f32) -> PV GEMM straight out of LDS -> f16 concat attn.
// Block: 128 threads = 4 waves. Dynamic LDS: 16 x LDS_P f32 strip + scratch.
// ---------------------------------------------------------------------------
__global__ __launch_bounds__(128) void fused_attn_kernel(
    const _Float16* __restrict__ qh, const _Float16* __restrict__ kh,
    const _Float16* __restrict__ vT, float* __restrict__ weights,
    _Float16* __restrict__ attnh) {
  extern __shared__ float smem[];
  float* sc  = smem;                 // [16][LDS_P] score strip
  float* red = smem + 16 * LDS_P;    // [8] cross-wave reduction scratch

  int hb  = blockIdx.y;              // b*H + h
  int bat = hb >> 4;
  int h   = hb & 15;
  const _Float16* q  = qh + (size_t)hb * SS * DEPTH;
  const _Float16* kk = kh + (size_t)hb * SS * DEPTH;
  const _Float16* vt = vT + (size_t)hb * DEPTH * SS;
  float* wout = weights + (size_t)hb * SS * SS;

  int t    = threadIdx.x;
  int w    = t >> 5;
  int lane = t & 31;
  int idx  = lane & 15;
  int hl   = lane >> 4;
  int row0 = blockIdx.x * 16;        // 16 query rows per block

  // ---- Phase 1: scores strip -> LDS (q A-fragments loaded once, reused) ----
  v16h a0 = load_a_f16(q, DEPTH, row0, 0);
  v16h a1 = load_a_f16(q, DEPTH, row0, 32);
  for (int j = 0; j < 32; ++j) {
    int n0 = (j * 4 + w) * 16;       // this wave's 16-col tile
    v8f acc = {};
    acc = wmma_f16(a0, load_bt_f16(kk, DEPTH, n0, 0), acc);
    acc = wmma_f16(a1, load_bt_f16(kk, DEPTH, n0, 32), acc);
#pragma unroll
    for (int i = 0; i < 8; ++i)
      sc[(size_t)(8 * hl + i) * LDS_P + n0 + idx] = acc[i] * 0.125f;
  }
  __syncthreads();

  // ---- Phase 2: softmax per row (all 128 threads cooperate per row) ----
  for (int r = 0; r < 16; ++r) {
    float* rp = sc + (size_t)r * LDS_P;
    float mx = -INFINITY;
#pragma unroll
    for (int i = 0; i < 16; ++i) mx = fmaxf(mx, rp[t + 128 * i]);
#pragma unroll
    for (int off = 16; off > 0; off >>= 1)
      mx = fmaxf(mx, __shfl_xor(mx, off, 32));
    if (lane == 0) red[w] = mx;
    __syncthreads();
    mx = fmaxf(fmaxf(red[0], red[1]), fmaxf(red[2], red[3]));

    float e[16];
    float sum = 0.0f;
#pragma unroll
    for (int i = 0; i < 16; ++i) {
      e[i] = __expf(rp[t + 128 * i] - mx);
      sum += e[i];
    }
#pragma unroll
    for (int off = 16; off > 0; off >>= 1)
      sum += __shfl_xor(sum, off, 32);
    if (lane == 0) red[4 + w] = sum;
    __syncthreads();
    sum = (red[4] + red[5]) + (red[6] + red[7]);
    float rs = 1.0f / sum;

#pragma unroll
    for (int i = 0; i < 16; ++i) {
      float pv = e[i] * rs;
      rp[t + 128 * i] = pv;                                   // for PV phase
      wout[(size_t)(row0 + r) * SS + t + 128 * i] = pv;       // the only write
    }
    __syncthreads();
  }

  // ---- Phase 3: PV GEMM from LDS. Wave w owns output cols [16w, 16w+16) ----
  int col0 = w * 16;
  v8f acc = {};
  for (int k0 = 0; k0 < SS; k0 += 32) {
    v16h af = load_a_lds_f32(sc, k0);
    v16h bf = load_bt_f16(vt, SS, col0, k0);
    acc = wmma_f16(af, bf, acc);
  }

  int d = h * DEPTH + col0 + idx;
#pragma unroll
  for (int i = 0; i < 8; ++i) {
    int s = row0 + 8 * hl + i;
    attnh[((size_t)(bat * SS + s)) * DD + d] = (_Float16)acc[i];
  }
}

// ---------------------------------------------------------------------------
// Kernel 3: out = attnh @ woT^T + bo   (M=4096, N=1024, K=1024), f32 output.
// ---------------------------------------------------------------------------
__global__ __launch_bounds__(128) void outproj_kernel(
    const _Float16* __restrict__ X, const _Float16* __restrict__ WT,
    const float* __restrict__ bias, float* __restrict__ out) {
  int w = threadIdx.x >> 5;
  int lane = threadIdx.x & 31;
  int idx = lane & 15;
  int hl = lane >> 4;
  int row0 = blockIdx.x * 64 + w * 16;
  int col0 = blockIdx.y * 64;

  v8f acc[4] = {};
  for (int k0 = 0; k0 < DD; k0 += 32) {
    v16h a = load_a_f16(X, DD, row0, k0);
#pragma unroll
    for (int tt = 0; tt < 4; ++tt) {
      v16h bf = load_bt_f16(WT, DD, col0 + 16 * tt, k0);
      acc[tt] = wmma_f16(a, bf, acc[tt]);
    }
  }

#pragma unroll
  for (int tt = 0; tt < 4; ++tt) {
    int c = col0 + 16 * tt + idx;
    float bv = bias[c];
#pragma unroll
    for (int i = 0; i < 8; ++i) {
      int r = row0 + 8 * hl + i;
      out[(size_t)r * DD + c] = acc[tt][i] + bv;
    }
  }
}

// ---------------------------------------------------------------------------
// Host-side launch
// ---------------------------------------------------------------------------
extern "C" void kernel_launch(void* const* d_in, const int* in_sizes, int n_in,
                              void* d_out, int out_size, void* d_ws, size_t ws_size,
                              hipStream_t stream) {
  const float* Q  = (const float*)d_in[0];
  const float* K  = (const float*)d_in[1];
  const float* V  = (const float*)d_in[2];
  const float* wq = (const float*)d_in[3];
  const float* bq = (const float*)d_in[4];
  const float* wk = (const float*)d_in[5];
  const float* bk = (const float*)d_in[6];
  const float* wv = (const float*)d_in[7];
  const float* bv = (const float*)d_in[8];
  const float* wo = (const float*)d_in[9];
  const float* bo = (const float*)d_in[10];

  float* out = (float*)d_out;                         // [2,2048,1024]
  float* weights = out + (size_t)BB * SS * DD;        // [2,16,2048,2048]

  // Workspace layout (f16 tensors), 64 MB total, all offsets 2MB-aligned.
  const size_t NBSD = (size_t)BB * SS * DD;   // 4,194,304 elements
  const size_t NW   = (size_t)DD * DD;        // 1,048,576 elements
  char* p = (char*)d_ws;
  _Float16* Qh   = (_Float16*)p;  p += NBSD * 2;
  _Float16* Kh   = (_Float16*)p;  p += NBSD * 2;
  _Float16* Vh   = (_Float16*)p;  p += NBSD * 2;
  _Float16* wqT  = (_Float16*)p;  p += NW * 2;
  _Float16* wkT  = (_Float16*)p;  p += NW * 2;
  _Float16* wvT  = (_Float16*)p;  p += NW * 2;
  _Float16* woT  = (_Float16*)p;  p += NW * 2;
  _Float16* qh   = (_Float16*)p;  p += NBSD * 2;   // [B,H,S,64]
  _Float16* kh   = (_Float16*)p;  p += NBSD * 2;   // [B,H,S,64]
  _Float16* vhT  = (_Float16*)p;  p += NBSD * 2;   // [B,H,64,S]
  _Float16* atnh = (_Float16*)p;  p += NBSD * 2;   // [B,S,1024] concat

  // 0) precision convert + weight transpose
  cvt_f16_kernel<<<(int)(NBSD / 256), 256, 0, stream>>>(Q, Qh, (int)NBSD);
  cvt_f16_kernel<<<(int)(NBSD / 256), 256, 0, stream>>>(K, Kh, (int)NBSD);
  cvt_f16_kernel<<<(int)(NBSD / 256), 256, 0, stream>>>(V, Vh, (int)NBSD);
  tcvt_f16_kernel<<<(int)(NW / 256), 256, 0, stream>>>(wq, wqT);
  tcvt_f16_kernel<<<(int)(NW / 256), 256, 0, stream>>>(wk, wkT);
  tcvt_f16_kernel<<<(int)(NW / 256), 256, 0, stream>>>(wv, wvT);
  tcvt_f16_kernel<<<(int)(NW / 256), 256, 0, stream>>>(wo, woT);

  // 1) QKV projections (M=4096, N=1024 tiled 64x64 per block)
  dim3 pg(BB * SS / 64, DD / 64);
  proj_kernel<<<pg, 128, 0, stream>>>(Qh, wqT, bq, qh, 0);
  proj_kernel<<<pg, 128, 0, stream>>>(Kh, wkT, bk, kh, 0);
  proj_kernel<<<pg, 128, 0, stream>>>(Vh, wvT, bv, vhT, 1);

  // 2) fused scores + softmax + PV  (weights tensor written exactly once)
  const int SMEM_BYTES = (16 * LDS_P + 8) * sizeof(float);  // 131,616 B
  fused_attn_kernel<<<dim3(SS / 16, BB * HH), 128, SMEM_BYTES, stream>>>(
      qh, kh, vhT, weights, atnh);

  // 3) output projection -> f32 d_out
  outproj_kernel<<<dim3(BB * SS / 64, DD / 64), 128, 0, stream>>>(
      atnh, woT, bo, out);
}
